// MSDeformAttn_15375982920297
// MI455X (gfx1250) — matrix operand order repaired and checked by reading, not compiled, not run on hardware
//
#include <hip/hip_runtime.h>
#include <hip/hip_bf16.h>
#include <math.h>

// ---------------------------------------------------------------------------
// MS-Deformable Attention for MI455X (gfx1250, wave32, WMMA, LDS-staged B)
// Pipeline: pack weights -> 3x f16-split WMMA GEMM -> softmax16 ->
//           bilinear sample -> WMMA GEMM
// ---------------------------------------------------------------------------

typedef __attribute__((ext_vector_type(16))) _Float16 v16h;
typedef __attribute__((ext_vector_type(8)))  float    v8f;

namespace cfg {
constexpr int Nb   = 2;       // batch
constexpr int Lq   = 16384;   // queries (128x128)
constexpr int C    = 256;     // d_model
constexpr int M    = 8;       // heads
constexpr int D    = 32;      // head dim
constexpr int Lin  = 21760;   // 128^2+64^2+32^2+16^2
constexpr int QW   = 128;     // query grid width (Lq = 128*128)
}

// ---------------------------------------------------------------------------
// Weight pre-pack: B[KxN] fp32  ->  WMMA B-fragment layout, f16 hi/lo split.
// Fragment (kb, ct): 32 lanes x (16 hi f16 + 16 lo f16) = 64B per lane
// (1024 f16 per fragment). Lane (n = lane&15, kblk = lane>>4), element e ->
// k = kb*32 + kblk*16 + e, col = ct*16 + n.
// ---------------------------------------------------------------------------
__global__ __launch_bounds__(256) void pack_weights(
    const float* __restrict__ B, _Float16* __restrict__ Wp, int Ncols, int K)
{
    const int idx   = blockIdx.x * blockDim.x + threadIdx.x;
    const int CT    = Ncols >> 4;
    const int total = (K >> 5) * CT * 32;
    if (idx >= total) return;

    const int lane = idx & 31;
    const int ct   = (idx >> 5) % CT;
    const int kb   = (idx >> 5) / CT;
    const int n    = lane & 15;
    const int kblk = lane >> 4;

    const float* bp = B + (size_t)(kb * 32 + kblk * 16) * Ncols + ct * 16 + n;
    _Float16*    w  = Wp + (size_t)idx * 32;
#pragma unroll
    for (int e = 0; e < 16; ++e) {
        const float f = bp[(size_t)e * Ncols];
        const _Float16 h = (_Float16)f;
        w[e]      = h;
        w[16 + e] = (_Float16)(f - (float)h);
    }
}

// ---------------------------------------------------------------------------
// GEMM  C[MxN] = A[MxK] @ B[KxN] + bias[N]
// Block = 256 threads = 8 waves = 128 rows x 64 cols. The packed B panel for
// this 64-col group over the whole K (64KB) is staged once into LDS and
// shared by all 8 waves; each wave then runs a 16x64 WMMA strip with A
// streamed from global (L2-resident) and B served from LDS (ds_load_b128).
// K % 32 == 0, Mrows % 16 == 0, Ncols % 64 == 0. grid = (Mrows/128, Ncols/64)
// ---------------------------------------------------------------------------
__global__ __launch_bounds__(256) void gemm_f16split(
    const float* __restrict__ A, const _Float16* __restrict__ Wp,
    const float* __restrict__ bias, float* __restrict__ Cout,
    int Mrows, int Ncols, int K)
{
    // 64 cols * K=256 * (hi+lo) f16 = 64KB  (K/32 * 4 fragments * 1024 f16)
    __shared__ __align__(16) _Float16 Bs[32 * 1024];

    const int lane = threadIdx.x & 31;
    const int wIb  = threadIdx.x >> 5;           // wave in block
    const int cg   = blockIdx.y;                 // 64-col group
    const int CT   = Ncols >> 4;

    // ---- cooperative stage of the packed B panel: global b128 -> LDS ----
    {
        const int totalUnits = (K >> 5) * 4 * 128;   // 16B units
        for (int u = threadIdx.x; u < totalUnits; u += 256) {
            const int f  = u >> 7;                   // fragment = kb*4 + t
            const int o  = u & 127;
            const int kb = f >> 2;
            const int t  = f & 3;
            const uint4 v = ((const uint4*)(Wp +
                ((size_t)(kb * CT + cg * 4 + t)) * 1024))[o];
            ((uint4*)Bs)[u] = v;
        }
    }
    __syncthreads();

    const int row0 = blockIdx.x * 128 + wIb * 16;
    if (row0 + 16 <= Mrows) {                    // wave-uniform guard
        const int col0 = cg << 6;
        const int m    = lane & 15;              // A row in tile / B,C column
        const int half = lane >> 4;

        v8f acc[4];
#pragma unroll
        for (int t = 0; t < 4; ++t) {
            const float bv = bias[col0 + t * 16 + m];
            acc[t] = (v8f){bv, bv, bv, bv, bv, bv, bv, bv};
        }

        const float* Arow = A + (size_t)(row0 + m) * K;

        for (int kb = 0; kb < (K >> 5); ++kb) {
            const int k0 = kb << 5;

            // ---- A fragment: one contiguous 16-float run -> 4x b128 ----
            const float4* a0 = (const float4*)(Arow + k0 + 16 * half);
            const float4* a1 = (const float4*)(Arow + k0 + 16 * half + 8);
            float af[16];
            *(float4*)(af + 0)  = a0[0];
            *(float4*)(af + 4)  = a0[1];
            *(float4*)(af + 8)  = a1[0];
            *(float4*)(af + 12) = a1[1];
            if (k0 + 32 < K)  // prefetch next A chunk (global_prefetch_b8)
                __builtin_prefetch(Arow + k0 + 32 + 8 * half, 0, 1);

            // ---- in-register f32 -> f16 hi/lo split of A ----
            v16h ah, al;
#pragma unroll
            for (int e = 0; e < 16; ++e) {
                const _Float16 h = (_Float16)af[e];
                ah[e] = h;
                al[e] = (_Float16)(af[e] - (float)h);
            }

            // ---- B fragments from LDS; D += Ah*Bh + Al*Bh + Ah*Bl ----
#pragma unroll
            for (int t = 0; t < 4; ++t) {
                const _Float16* bp = Bs + ((size_t)(kb * 4 + t) * 32 + lane) * 32;
                const v16h bh = *(const v16h*)(bp);
                const v16h bl = *(const v16h*)(bp + 16);
                acc[t] = __builtin_amdgcn_wmma_f32_16x16x32_f16(
                    false, ah, false, bh, (short)0, acc[t], false, false);
                acc[t] = __builtin_amdgcn_wmma_f32_16x16x32_f16(
                    false, al, false, bh, (short)0, acc[t], false, false);
                acc[t] = __builtin_amdgcn_wmma_f32_16x16x32_f16(
                    false, ah, false, bl, (short)0, acc[t], false, false);
            }
        }

        // ---- store: row = r + 8*half, col = col0 + t*16 + m ----
#pragma unroll
        for (int t = 0; t < 4; ++t) {
            float* cp = Cout + (size_t)(row0 + 8 * half) * Ncols
                             + col0 + t * 16 + m;
#pragma unroll
            for (int r = 0; r < 8; ++r)
                cp[(size_t)r * Ncols] = acc[t][r];
        }
    }
}

// ---------------------------------------------------------------------------
// In-place softmax over contiguous groups of 16 (one thread per group)
// ---------------------------------------------------------------------------
__global__ __launch_bounds__(256) void softmax16(float* __restrict__ a, int rows)
{
    const int i = blockIdx.x * blockDim.x + threadIdx.x;
    if (i >= rows) return;
    float* p = a + (size_t)i * 16;
    float e[16];
    float mx = p[0];
#pragma unroll
    for (int j = 1; j < 16; ++j) mx = fmaxf(mx, p[j]);
    float s = 0.f;
#pragma unroll
    for (int j = 0; j < 16; ++j) { e[j] = expf(p[j] - mx); s += e[j]; }
    const float inv = 1.0f / s;
#pragma unroll
    for (int j = 0; j < 16; ++j) p[j] = e[j] * inv;
}

// ---------------------------------------------------------------------------
// Bilinear deformable sampling + attention aggregation.
// One wave per (n, q, m); lane = channel d in [0,32). All location math is
// wave-uniform; the 4 gathers per (l,p) are 128B coalesced across lanes.
//   value: [N, Lin, M, D]   off: [N, Lq, M, L, P, 2]
//   attn : [N, Lq, M, 16]   mid: [N, Lq, M*D]
// ---------------------------------------------------------------------------
__global__ __launch_bounds__(256) void msda_sample(
    const float* __restrict__ value, const float* __restrict__ off,
    const float* __restrict__ attn,  float* __restrict__ mid)
{
    using namespace cfg;
    const int lane = threadIdx.x & 31;
    const int wid  = blockIdx.x * 8 + (threadIdx.x >> 5);
    const int m  = wid & 7;
    const int nq = wid >> 3;          // n*Lq + q
    const int q  = nq & (Lq - 1);
    const int n  = nq >> 14;

    const float refx = ((q & (QW - 1)) + 0.5f) * (1.0f / QW);
    const float refy = ((q >> 7) + 0.5f) * (1.0f / QW);

    const float* offp  = off  + ((size_t)nq * M + m) * 32;  // L*P*2 = 32
    const float* attnp = attn + ((size_t)nq * M + m) * 16;  // L*P   = 16

    const int Hs[4]     = {128, 64, 32, 16};
    const int starts[4] = {0, 16384, 20480, 21504};

    float acc = 0.0f;
#pragma unroll
    for (int l = 0; l < 4; ++l) {
        const int H = Hs[l];
        const int W = H;               // square levels
        const float* vb = value + ((size_t)(n * Lin + starts[l]) * M + m) * D + lane;
#pragma unroll
        for (int p = 0; p < 4; ++p) {
            const float ox = offp[l * 8 + p * 2 + 0];
            const float oy = offp[l * 8 + p * 2 + 1];
            const float aw = attnp[l * 4 + p];
            const float x = refx * W + ox - 0.5f;
            const float y = refy * H + oy - 0.5f;
            const float x0f = floorf(x), y0f = floorf(y);
            const int x0 = (int)x0f, y0 = (int)y0f;
            const float wx1 = x - x0f, wy1 = y - y0f;
            const float wx0 = 1.0f - wx1, wy0 = 1.0f - wy1;

            float s = 0.0f;
#pragma unroll
            for (int cy = 0; cy < 2; ++cy) {
#pragma unroll
                for (int cx = 0; cx < 2; ++cx) {
                    const int yy = y0 + cy, xx = x0 + cx;
                    const float wgt = (cy ? wy1 : wy0) * (cx ? wx1 : wx0);
                    if (xx >= 0 && xx < W && yy >= 0 && yy < H) {
                        s += wgt * vb[(size_t)(yy * W + xx) * (M * D)];
                    }
                }
            }
            acc += aw * s;
        }
    }
    mid[(size_t)nq * C + m * D + lane] = acc;
}

// ---------------------------------------------------------------------------
// Host launcher
// ---------------------------------------------------------------------------
extern "C" void kernel_launch(void* const* d_in, const int* in_sizes, int n_in,
                              void* d_out, int out_size, void* d_ws, size_t ws_size,
                              hipStream_t stream)
{
    using namespace cfg;
    const float* query         = (const float*)d_in[0];
    // d_in[1] reference_points: unused by the reference computation
    const float* input_flatten = (const float*)d_in[2];
    // d_in[3] spatial_shapes, d_in[4] level_start_index: compile-time constants
    const float* W_off  = (const float*)d_in[5];
    const float* b_off  = (const float*)d_in[6];
    const float* W_attn = (const float*)d_in[7];
    const float* b_attn = (const float*)d_in[8];
    const float* W_val  = (const float*)d_in[9];
    const float* b_val  = (const float*)d_in[10];
    const float* W_out  = (const float*)d_in[11];
    const float* b_out  = (const float*)d_in[12];

    // workspace layout (float units)
    float* ws    = (float*)d_ws;
    float* value = ws;                                   // 43520*256
    float* offb  = value + (size_t)Nb * Lin * C;         // 32768*256
    float* attnb = offb  + (size_t)Nb * Lq  * C;         // 32768*128
    float* midb  = attnb + (size_t)Nb * Lq  * (C / 2);   // 32768*256
    float* wtail = midb  + (size_t)Nb * Lq  * C;
    // packed weights: K*N f16 hi + K*N f16 lo  ==  K*N floats of space
    _Float16* Wp_val  = (_Float16*)wtail;                       // 256*256*2 f16
    _Float16* Wp_off  = Wp_val  + (size_t)C * C * 2;
    _Float16* Wp_attn = Wp_off  + (size_t)C * C * 2;
    _Float16* Wp_out  = Wp_attn + (size_t)C * (C / 2) * 2;
    float* out = (float*)d_out;

    auto pack = [&](const float* B, _Float16* Wp, int Nc) {
        const int total = (C / 32) * (Nc / 16) * 32;
        hipLaunchKernelGGL(pack_weights, dim3((total + 255) / 256), dim3(256),
                           0, stream, B, Wp, Nc, C);
    };
    auto gemm = [&](const float* A, const _Float16* Wp, const float* bias,
                    float* Cp, int Mr, int Nc) {
        hipLaunchKernelGGL(gemm_f16split,
                           dim3((Mr + 127) / 128, Nc / 64), dim3(256),
                           0, stream, A, Wp, bias, Cp, Mr, Nc, C);
    };

    pack(W_val,  Wp_val,  C);
    pack(W_off,  Wp_off,  C);
    pack(W_attn, Wp_attn, C / 2);
    pack(W_out,  Wp_out,  C);

    gemm(input_flatten, Wp_val,  b_val,  value, Nb * Lin, C);      // value proj
    gemm(query,         Wp_off,  b_off,  offb,  Nb * Lq,  C);      // offsets
    gemm(query,         Wp_attn, b_attn, attnb, Nb * Lq,  C / 2);  // attn logits

    const int rows = Nb * Lq * M;
    hipLaunchKernelGGL(softmax16, dim3((rows + 255) / 256), dim3(256), 0, stream,
                       attnb, rows);

    hipLaunchKernelGGL(msda_sample, dim3(rows / 8), dim3(256), 0, stream,
                       value, offb, attnb, midb);

    gemm(midb, Wp_out, b_out, out, Nb * Lq, C);                    // output proj
}
